// SpatialAttentionBlock_69767448756879
// MI455X (gfx1250) — matrix operand order repaired and checked by reading, not compiled, not run on hardware
//
#include <hip/hip_runtime.h>
#include <hip/hip_bf16.h>

// ---------------------------------------------------------------------------
// SpatialAttentionBlock for MI455X (gfx1250, wave32, WMMA bf16, async->LDS)
//   a: [4,256,64,64] f32 ; b_w/c_w: [32,256]; d_w: [256,256]; alpha scalar
//   out = alpha * (V @ softmax(Q K^T)^T) + a
// Flash style; un-normalized exp accumulation (scores ~N(0,32), overflow-free),
// triple-buffered async global->LDS staging of K/V tiles.
// ---------------------------------------------------------------------------

#define HW   4096   // 64*64 tokens
#define CCH  256    // channels
#define NB   4      // batch

typedef __bf16 bf16;
typedef __bf16 v16bf __attribute__((ext_vector_type(16)));
typedef float  v8f   __attribute__((ext_vector_type(8)));

union ABfrag { v16bf v; unsigned u[8]; };
union CDfrag { v8f   v; float    f[8]; };

// A-matrix 16x32 bf16 per-lane K index (contiguous, matches B): lanes 0-15 hold
// K={0..15}, lanes 16-31 hold K={16..31}, in u32 pairs.
__device__ __forceinline__ int a_klocal(int vv, int h) {
    return 16 * h + 2 * vv;
}

// LDS byte offset of a __shared__ object (generic address low 32 bits)
__device__ __forceinline__ unsigned lds_off(const void* p) {
    return (unsigned)(size_t)p;
}

// gfx1250 async global -> LDS copies (ASYNCcnt-tracked)
__device__ __forceinline__ void async_b64(unsigned dst, const void* src) {
    asm volatile("global_load_async_to_lds_b64 %0, %1, off"
                 :: "v"(dst), "v"(src) : "memory");
}
__device__ __forceinline__ void async_b128(unsigned dst, const void* src) {
    asm volatile("global_load_async_to_lds_b128 %0, %1, off"
                 :: "v"(dst), "v"(src) : "memory");
}
__device__ __forceinline__ void wait_async_le5() {   // drain oldest tile (5 ops)
    asm volatile("s_wait_asynccnt 0x5" ::: "memory");
}

// ---------------------------------------------------------------------------
// Kernel 0: one-shot weight conversion f32 -> bf16 into wb[320][256]
//   rows 0..31 = b_w, 32..63 = c_w, 64..319 = d_w
// ---------------------------------------------------------------------------
__global__ __launch_bounds__(256)
void prep_w_kernel(const float* __restrict__ b_w, const float* __restrict__ c_w,
                   const float* __restrict__ d_w, bf16* __restrict__ wb)
{
    const int idx = blockIdx.x * 256 + threadIdx.x;   // 0 .. 320*256-1
    const int o = idx >> 8, c = idx & 255;
    const float* src = (o < 32) ? (b_w + (size_t)o * CCH)
                     : (o < 64) ? (c_w + (size_t)(o - 32) * CCH)
                                : (d_w + (size_t)(o - 64) * CCH);
    wb[idx] = (bf16)src[c];
}

// ---------------------------------------------------------------------------
// Kernel 1: fused projections.  P[n][i][o] = sum_c a[n][c][i] * wb[o][c] + bias
//   o in [0,32)   -> Q  [n][i][32]   (bf16)
//   o in [32,64)  -> K  [n][i][32]   (bf16)
//   o in [64,320) -> V  [n][c][4096] (bf16, channel-major like "d")
// Block: 128 threads (4 waves); wave computes 16(i) x 16(o) via 8 WMMAs.
// ---------------------------------------------------------------------------
__global__ __launch_bounds__(128)
void proj_kernel(const float* __restrict__ a, const bf16* __restrict__ wb,
                 const float* __restrict__ b_b, const float* __restrict__ c_b,
                 const float* __restrict__ d_b,
                 bf16* __restrict__ q, bf16* __restrict__ k, bf16* __restrict__ v)
{
    __shared__ bf16 atile[16 * CCH];   // [i 0..15][c 0..255], bf16

    const int n    = blockIdx.z;
    const int i0   = blockIdx.x * 16;
    const int wave = threadIdx.x >> 5;
    const int lane = threadIdx.x & 31;
    const int o0   = blockIdx.y * 64 + wave * 16;

    // ---- stage a[n][:, i0..i0+15] into LDS as bf16, layout [i][c] ----
    const float* ga = a + (size_t)n * CCH * HW + i0;
    for (int t = threadIdx.x; t < 1024; t += 128) {
        const int c = t >> 2, seg = t & 3;                 // 4 float4 per c-row
        const float4 f4 = *(const float4*)(ga + (size_t)c * HW + seg * 4);
        atile[(seg * 4 + 0) * CCH + c] = (bf16)f4.x;
        atile[(seg * 4 + 1) * CCH + c] = (bf16)f4.y;
        atile[(seg * 4 + 2) * CCH + c] = (bf16)f4.z;
        atile[(seg * 4 + 3) * CCH + c] = (bf16)f4.w;
    }
    __syncthreads();

    // ---- bias select (o-tiles never straddle regions) ----
    const float* bias; int obase;
    if (o0 < 32)      { bias = b_b; obase = o0;      }
    else if (o0 < 64) { bias = c_b; obase = o0 - 32; }
    else              { bias = d_b; obase = o0 - 64; }

    const int h   = lane >> 4;
    const int l15 = lane & 15;
    const int oc  = obase + l15;                 // column within region
    const float bv = bias[oc];
    const bf16* wrow = wb + (size_t)(o0 + l15) * CCH + 16 * h;  // bf16 weights

    CDfrag acc;
    #pragma unroll
    for (int r = 0; r < 8; ++r) acc.f[r] = 0.0f;

    #pragma unroll
    for (int kc = 0; kc < CCH; kc += 32) {
        ABfrag af;                               // A: 16(i) x 32(c) bf16
        #pragma unroll
        for (int vv = 0; vv < 8; ++vv)
            af.u[vv] = *(const unsigned*)&atile[l15 * CCH + kc + a_klocal(vv, h)];

        ABfrag bfrag;                            // B: 32(c) x 16(o), K = 16h+e
        const bf16* wp = wrow + kc;
        #pragma unroll
        for (int e = 0; e < 8; ++e)
            bfrag.u[e] = *(const unsigned*)(wp + 2 * e);

        acc.v = __builtin_amdgcn_wmma_f32_16x16x32_bf16(
                    false, af.v, false, bfrag.v, (short)0, acc.v, false, false);
    }

    #pragma unroll
    for (int r = 0; r < 8; ++r) acc.f[r] += bv;

    // ---- store (D layout: VGPR r -> row i0+r+8h, col = oc) ----
    if (o0 < 64) {                               // Q / K : [n][i][32]
        bf16* dst = ((o0 < 32) ? q : k) + (size_t)n * HW * 32;
        #pragma unroll
        for (int r = 0; r < 8; ++r)
            dst[(size_t)(i0 + r + 8 * h) * 32 + oc] = (bf16)acc.f[r];
    } else {                                     // V : [n][c][HW], i contiguous
        bf16* dst = v + ((size_t)n * CCH + oc) * HW + i0 + 8 * h;
        #pragma unroll
        for (int rr = 0; rr < 4; ++rr) {
            union { bf16 b[2]; unsigned u; } p;
            p.b[0] = (bf16)acc.f[2 * rr];
            p.b[1] = (bf16)acc.f[2 * rr + 1];
            *(unsigned*)(dst + 2 * rr) = p.u;
        }
    }
}

// ---------------------------------------------------------------------------
// Kernel 2: flash attention, un-normalized exp, triple-buffered async K/V.
// Block = 256 threads = 8 waves; block owns 32 query rows; wave (it=w>>2,
// cg=w&3) owns i-tile `it` and V columns [cg*64, cg*64+64).
// One barrier + one s_wait_asynccnt per 32-key step.
// ---------------------------------------------------------------------------
__global__ __launch_bounds__(256)
void attn_kernel(const float* __restrict__ a, const float* __restrict__ alpha_p,
                 const bf16* __restrict__ q, const bf16* __restrict__ kk,
                 const bf16* __restrict__ v, float* __restrict__ out)
{
    __shared__ bf16 klds[3][32 * 32];            //  3 x  2 KB : K tile [j][32c]
    __shared__ bf16 vlds[3][256 * 32];           //  3 x 16 KB : V tile [c][32j]
    __shared__ bf16 pbuf[8][16 * 32];            //       8 KB : per-wave P transpose

    const int n    = blockIdx.y;
    const int i0   = blockIdx.x * 32;
    const int tid  = threadIdx.x;
    const int wave = tid >> 5;
    const int lane = tid & 31;
    const int h = lane >> 4, l15 = lane & 15;
    const int it = wave >> 2;                    // which 16-row i-tile
    const int cg = wave & 3;                     // which 64-col value group
    const int ibase = i0 + it * 16;
    const float alpha = alpha_p[0];

    const bf16* qn = q  + (size_t)n * HW * 32;
    const bf16* kn = kk + (size_t)n * HW * 32;
    const bf16* vn = v  + (size_t)n * CCH * HW;

    // async-copy addressing: each thread moves 8B of K + 64B of V per tile
    const int kj = tid >> 3, kpart = tid & 7;
    const bf16* vsrc_row = vn + (size_t)tid * HW;
    unsigned kdst[3], vdst[3];
    #pragma unroll
    for (int b = 0; b < 3; ++b) {
        kdst[b] = lds_off(&klds[b][kj * 32 + kpart * 4]);
        vdst[b] = lds_off(&vlds[b][tid * 32]);
    }
    auto issue_tile = [&](int b, int j0) {       // exactly 5 async ops
        async_b64(kdst[b], kn + (size_t)(j0 + kj) * 32 + kpart * 4);
        #pragma unroll
        for (int seg = 0; seg < 4; ++seg)
            async_b128(vdst[b] + seg * 16, vsrc_row + j0 + seg * 8);
    };

    // Q fragment (A-layout), resident for the whole kernel
    ABfrag qf;
    #pragma unroll
    for (int vv = 0; vv < 8; ++vv)
        qf.u[vv] = *(const unsigned*)(qn + (size_t)(ibase + l15) * 32 + a_klocal(vv, h));

    CDfrag acc[4];
    #pragma unroll
    for (int ct = 0; ct < 4; ++ct)
        #pragma unroll
        for (int r = 0; r < 8; ++r) acc[ct].f[r] = 0.0f;

    CDfrag zf;
    #pragma unroll
    for (int r = 0; r < 8; ++r) zf.f[r] = 0.0f;

    // per-lane PARTIAL row sums (this lane's columns only); reduced once at end
    float lrow[8];
    #pragma unroll
    for (int r = 0; r < 8; ++r) lrow[r] = 0.0f;

    bf16* myp = pbuf[wave];

    issue_tile(0, 0);
    issue_tile(1, 32);

    for (int jb = 0; jb < HW / 32; ++jb) {
        wait_async_le5();                        // tile jb resident (in-order)
        __syncthreads();                         // all waves' data visible; all
                                                 // waves done with tile jb-1
        const int jn = (jb + 2 < HW / 32) ? (jb + 2) : (HW / 32 - 1);
        issue_tile((jb + 2) % 3, jn * 32);       // buffer last read at tile jb-1

        const bf16* kT = klds[jb % 3];
        const bf16* vT = vlds[jb % 3];

        // ---- scores: S[i][j] for two 16-col j-tiles ----
        CDfrag s[2];
        #pragma unroll
        for (int jt = 0; jt < 2; ++jt) {
            ABfrag kf;                            // B: K(c)=16h+e, col = key j
            const bf16* kp = &kT[(16 * jt + l15) * 32 + 16 * h];
            #pragma unroll
            for (int e = 0; e < 8; ++e)
                kf.u[e] = *(const unsigned*)(kp + 2 * e);
            s[jt].v = __builtin_amdgcn_wmma_f32_16x16x32_bf16(
                          false, qf.v, false, kf.v, (short)0, zf.v, false, false);
        }

        // ---- V fragments ----
        ABfrag vf[4];
        #pragma unroll
        for (int ct = 0; ct < 4; ++ct) {
            const bf16* vp = &vT[(cg * 64 + ct * 16 + l15) * 32 + 16 * h];
            #pragma unroll
            for (int e = 0; e < 8; ++e)
                vf[ct].u[e] = *(const unsigned*)(vp + 2 * e);
        }

        // ---- un-normalized softmax numerator + P transpose to A-layout ----
        #pragma unroll
        for (int r = 0; r < 8; ++r) {
            const float p0 = __expf(s[0].f[r]);
            const float p1 = __expf(s[1].f[r]);
            lrow[r] += p0 + p1;                  // per-lane partial row sum
            myp[(r + 8 * h) * 32 + l15]      = (bf16)p0;
            myp[(r + 8 * h) * 32 + 16 + l15] = (bf16)p1;
        }

        ABfrag pf;                               // same-wave LDS ops are in-order
        #pragma unroll
        for (int vv = 0; vv < 8; ++vv)
            pf.u[vv] = *(const unsigned*)&myp[l15 * 32 + a_klocal(vv, h)];

        // ---- accumulate O += P @ V ----
        #pragma unroll
        for (int ct = 0; ct < 4; ++ct)
            acc[ct].v = __builtin_amdgcn_wmma_f32_16x16x32_bf16(
                            false, pf.v, false, vf[ct].v, (short)0, acc[ct].v, false, false);
    }

    // ---- one final cross-lane row-sum reduction (width 16 = C/D half) ----
    float inv[8];
    #pragma unroll
    for (int r = 0; r < 8; ++r) {
        float rs = lrow[r];
        #pragma unroll
        for (int d = 1; d < 16; d <<= 1)
            rs += __shfl_xor(rs, d, 16);
        inv[r] = 1.0f / rs;
    }

    // ---- normalize, alpha-scale, residual add, store ----
    const float* an = a   + (size_t)n * CCH * HW;
    float*       on = (float*)out + (size_t)n * CCH * HW;
    #pragma unroll
    for (int ct = 0; ct < 4; ++ct) {
        const int c = cg * 64 + ct * 16 + l15;
        #pragma unroll
        for (int r = 0; r < 8; ++r) {
            const int i = ibase + r + 8 * h;
            const size_t idx = (size_t)c * HW + i;
            on[idx] = alpha * acc[ct].f[r] * inv[r] + an[idx];
        }
    }
}

// ---------------------------------------------------------------------------
extern "C" void kernel_launch(void* const* d_in, const int* in_sizes, int n_in,
                              void* d_out, int out_size, void* d_ws, size_t ws_size,
                              hipStream_t stream) {
    const float* a   = (const float*)d_in[0];
    const float* b_w = (const float*)d_in[1];
    const float* b_b = (const float*)d_in[2];
    const float* c_w = (const float*)d_in[3];
    const float* c_b = (const float*)d_in[4];
    const float* d_w = (const float*)d_in[5];
    const float* d_b = (const float*)d_in[6];
    const float* alp = (const float*)d_in[7];

    bf16* q  = (bf16*)d_ws;                      //  1 MB : [4][4096][32]
    bf16* k  = q  + (size_t)NB * HW * 32;        //  1 MB : [4][4096][32]
    bf16* v  = k  + (size_t)NB * HW * 32;        //  8 MB : [4][256][4096]
    bf16* wb = v  + (size_t)NB * CCH * HW;       // 160KB : [320][256]

    // one-shot bf16 weight conversion
    prep_w_kernel<<<dim3(320), 256, 0, stream>>>(b_w, c_w, d_w, wb);

    // projections: grid (i-tiles=256, o-blocks=5 (=320/64), batch=4)
    proj_kernel<<<dim3(HW / 16, 320 / 64, NB), 128, 0, stream>>>(
        a, wb, b_b, c_b, d_b, q, k, v);

    // flash attention: grid (query blocks of 32 rows = 128, batch = 4)
    attn_kernel<<<dim3(HW / 32, NB), 256, 0, stream>>>(
        a, alp, q, k, v, (float*)d_out);
}
